// GaborFilter_77068893160155
// MI455X (gfx1250) — matrix operand order repaired and checked by reading, not compiled, not run on hardware
//
#include <hip/hip_runtime.h>
#include <hip/hip_bf16.h>

typedef __attribute__((ext_vector_type(16))) _Float16 v16h;
typedef __attribute__((ext_vector_type(8)))  float    v8f;

#define IMG_H 512
#define IMG_W 512
#define NBATCH 2
#define NF 180
#define KH 17
#define MT 12            // 192 padded filters / 16
#define TILE_X 128       // pixels per block (8 waves x 16)
#define PATCH_W 144      // 128 + 16 halo columns
#define PATCH_PITCH 160  // LDS row pitch
#define PI_F 3.14159265358979323846f

// ---------------- wave32 xor-16 exchange via ds_swizzle (SWAPX16) -----------
__device__ __forceinline__ int swap16_i(int v) {
    // group-of-32 mode: and_mask=0x1f, or_mask=0, xor_mask=0x10 -> offset 0x401f
    return __builtin_amdgcn_ds_swizzle(v, 0x401f);
}
__device__ __forceinline__ float swap16_f(float v) {
    return __int_as_float(__builtin_amdgcn_ds_swizzle(__float_as_int(v), 0x401f));
}

// ---------------- stage 0: swizzle Gabor weights into WMMA A-fragment order -
// A-matrix 16x32 f16 layout (ISA 7.12.2): lane L holds row M = L%16;
// half j of v16h maps to K = (j<8 ? 0 : 16) + (L<16 ? 0 : 8) + (j&7).
// Stored as wA[(mt*17 + ky)*32 + lane] : v16h  (K dim = kx, padded 17->32)
__global__ __launch_bounds__(256)
void prep_weights_kernel(const float* __restrict__ w, _Float16* __restrict__ wA) {
    int t = blockIdx.x * blockDim.x + threadIdx.x;
    const int total = MT * KH * 32;
    if (t >= total) return;
    int lane = t & 31;
    int ky   = (t >> 5) % KH;
    int mt   = t / (KH * 32);
    int M    = mt * 16 + (lane & 15);
    int kb   = (lane & 16) ? 8 : 0;
    v16h o;
#pragma unroll
    for (int j = 0; j < 16; ++j) {
        int kx = ((j >> 3) << 4) + kb + (j & 7);
        float v = 0.0f;
        if (M < NF && kx < KH) v = w[(M * KH + ky) * KH + kx];
        o[j] = (_Float16)v;
    }
    ((v16h*)wA)[t] = o;
}

// ---------------- stage 1: RGB -> gray ------------------------------------
__global__ __launch_bounds__(256)
void gray_kernel(const float* __restrict__ x, float* __restrict__ gray) {
    int idx = blockIdx.x * blockDim.x + threadIdx.x;
    const int npix = NBATCH * IMG_H * IMG_W;
    if (idx >= npix) return;
    int b = idx / (IMG_H * IMG_W);
    int p = idx - b * (IMG_H * IMG_W);
    const float* base = x + (size_t)b * 3 * IMG_H * IMG_W + p;
    gray[idx] = 0.2989f * base[0]
              + 0.587f  * base[IMG_H * IMG_W]
              + 0.114f  * base[2 * IMG_H * IMG_W];
}

// ---------------- stage 2: separable Gaussian blurs (edge-replicate) -------
__global__ __launch_bounds__(256)
void blurV_kernel(const float* __restrict__ in, float* __restrict__ out,
                  int r, float sigma) {
    __shared__ float taps[81];
    int t = threadIdx.x;
    if (t <= 2 * r) {
        float d = (float)(t - r) / sigma;
        taps[t] = expf(-0.5f * d * d);
    }
    __syncthreads();
    float s = 0.0f;
    for (int i = 0; i <= 2 * r; ++i) s += taps[i];
    float inv = 1.0f / s;

    int idx = blockIdx.x * blockDim.x + threadIdx.x;
    const int npix = NBATCH * IMG_H * IMG_W;
    if (idx >= npix) return;
    int b = idx / (IMG_H * IMG_W);
    int p = idx - b * (IMG_H * IMG_W);
    int y = p / IMG_W, xx = p - y * IMG_W;
    const float* img = in + (size_t)b * IMG_H * IMG_W;
    float acc = 0.0f;
    for (int i = 0; i <= 2 * r; ++i) {
        int yy = y + i - r;
        yy = yy < 0 ? 0 : (yy > IMG_H - 1 ? IMG_H - 1 : yy);
        acc += taps[i] * img[yy * IMG_W + xx];
    }
    out[idx] = acc * inv;
}

__global__ __launch_bounds__(256)
void blurH_kernel(const float* __restrict__ in, float* __restrict__ out,
                  int r, float sigma, int doSub, const float* __restrict__ sub) {
    __shared__ float taps[81];
    int t = threadIdx.x;
    if (t <= 2 * r) {
        float d = (float)(t - r) / sigma;
        taps[t] = expf(-0.5f * d * d);
    }
    __syncthreads();
    float s = 0.0f;
    for (int i = 0; i <= 2 * r; ++i) s += taps[i];
    float inv = 1.0f / s;

    int idx = blockIdx.x * blockDim.x + threadIdx.x;
    const int npix = NBATCH * IMG_H * IMG_W;
    if (idx >= npix) return;
    int b = idx / (IMG_H * IMG_W);
    int p = idx - b * (IMG_H * IMG_W);
    int y = p / IMG_W, xx = p - y * IMG_W;
    const float* row = in + ((size_t)b * IMG_H + y) * IMG_W;
    float acc = 0.0f;
    for (int i = 0; i <= 2 * r; ++i) {
        int c = xx + i - r;
        c = c < 0 ? 0 : (c > IMG_W - 1 ? IMG_W - 1 : c);
        acc += taps[i] * row[c];
    }
    acc *= inv;
    out[idx] = doSub ? (sub[idx] - acc) : acc;
}

// ---------------- stage 3: WMMA Gabor bank + argmax + confidence -----------
// Block = 256 threads (8 waves). Block tile = 128 pixels of one image row.
// Wave handles 16 pixels (N) x 192 filters (12 M-tiles), K split per ky row.
// A fragments use a 4-deep rotating buffer (loads stay >=3 fragments ahead of
// consumption); B fragments double-buffer across ky.
__global__ __launch_bounds__(256)
void gabor_wmma_kernel(const float* __restrict__ filtered,
                       const _Float16* __restrict__ wA,
                       float* __restrict__ outOri,
                       float* __restrict__ outConf) {
    __shared__ _Float16 patch[KH * PATCH_PITCH];

    const int b  = blockIdx.z;
    const int y  = blockIdx.y;
    const int x0 = blockIdx.x * TILE_X;
    const int tid = threadIdx.x;

    // ---- stage the 17 x 144 input patch (zero padded) into LDS as f16 ----
    const float* img = filtered + (size_t)b * IMG_H * IMG_W;
    for (int i = tid; i < KH * PATCH_W; i += 256) {
        int row = i / PATCH_W;
        int c   = i - row * PATCH_W;
        int gx  = x0 - 8 + c;
        int gy  = y - 8 + row;
        float v = 0.0f;
        if (gx >= 0 && gx < IMG_W && gy >= 0 && gy < IMG_H)
            v = img[gy * IMG_W + gx];
        patch[row * PATCH_PITCH + c] = (_Float16)v;
    }
    __syncthreads();

    const int wave  = tid >> 5;
    const int lane  = tid & 31;
    const int n     = lane & 15;
    const int kbase = (lane & 16) ? 16 : 0;
    const int colb  = wave * 16 + n;

    // B-fragment builder (32x16 K x N): element j holds K = (j<8 ? 0 : 16) +
    // kbase/2 + (j&7) of pixel n; K dim = kx (17 valid, rest zero).
    auto buildB = [&](int ky) -> v16h {
        const _Float16* prow = &patch[ky * PATCH_PITCH + colb];
        v16h f;
#pragma unroll
        for (int j = 0; j < 16; ++j) {
            int kx = ((j >> 3) << 4) + (kbase >> 1) + (j & 7);
            f[j] = (kx < KH) ? prow[kx] : (_Float16)0.0f;
        }
        return f;
    };

    v8f acc[MT];
#pragma unroll
    for (int i = 0; i < MT; ++i)
        acc[i] = (v8f){0.f, 0.f, 0.f, 0.f, 0.f, 0.f, 0.f, 0.f};

    const v16h* wav = (const v16h*)wA + lane;

    v16h bfrag = buildB(0);

    // 4-deep rotating A-fragment pipeline: slot (mt & 3) holds fragment mt,
    // reloaded with fragment mt+4 right after consumption.
    v16h aBuf[4];
#pragma unroll
    for (int i = 0; i < 4; ++i)
        aBuf[i] = wav[(i * KH + 0) * 32];

#pragma unroll 1
    for (int ky = 0; ky < KH; ++ky) {
        const int kyn = (ky + 1 < KH) ? ky + 1 : KH - 1;
        // pull next ky's weight slab (12 KB) toward WGP$ while we compute
        __builtin_prefetch((const void*)(wav + kyn * 32), 0, 3);
        // issue DS loads for next ky's B fragment ahead of the WMMAs
        v16h bNext = buildB(kyn);
#pragma unroll
        for (int mt = 0; mt < MT; ++mt) {
            acc[mt] = __builtin_amdgcn_wmma_f32_16x16x32_f16(
                false, aBuf[mt & 3], false, bfrag, (short)0, acc[mt],
                false, false);
            const int nmt = mt + 4;
            aBuf[mt & 3] = (nmt < MT) ? wav[(nmt * KH + ky) * 32]
                                      : wav[((nmt - MT) * KH + kyn) * 32];
        }
        bfrag = bNext;
    }

    // ---- per-lane reduction over its 96 filter responses ----
    // C/D layout: VGPR r of acc[mt] holds M = mt*16 + r + (lane<16 ? 0 : 8)
    const int mbase = (lane & 16) ? 8 : 0;
    float sumAbs = 0.0f, best = -1.0f;
    int bestIdx = 0;
#pragma unroll
    for (int mt = 0; mt < MT; ++mt) {
#pragma unroll
        for (int r = 0; r < 8; ++r) {
            int M = mt * 16 + mbase + r;
            if (M < NF) {
                float v = fabsf(acc[mt][r]);
                sumAbs += v;
                if (v > best) { best = v; bestIdx = M; }
            }
        }
    }

    // combine with partner lane (same pixel, other M half) via ds_swizzle
    float ob = swap16_f(best);
    int   oi = swap16_i(bestIdx);
    sumAbs  += swap16_f(sumAbs);
    if (ob > best || (ob == best && oi < bestIdx)) { best = ob; bestIdx = oi; }

    float ori = (float)bestIdx * (255.0f / (float)NF);
    float rad = ori * (PI_F / (float)NF);

    // second pass: sum dist^2 * |response|
    float conf = 0.0f;
#pragma unroll
    for (int mt = 0; mt < MT; ++mt) {
#pragma unroll
        for (int r = 0; r < 8; ++r) {
            int M = mt * 16 + mbase + r;
            if (M < NF) {
                float bin = (float)M * (PI_F / (float)NF);
                float d0 = fabsf(rad - bin);
                float d  = fminf(d0, fminf(fabsf(rad - bin - PI_F),
                                           fabsf(rad - bin + PI_F)));
                conf += d * d * fabsf(acc[mt][r]);
            }
        }
    }
    conf += swap16_f(conf);

    if (lane < 16) {
        int xx = x0 + wave * 16 + n;
        size_t p = ((size_t)b * IMG_H + y) * IMG_W + xx;
        outOri[p]  = ori;
        outConf[p] = conf / sumAbs;
    }
}

// ---------------- launcher --------------------------------------------------
extern "C" void kernel_launch(void* const* d_in, const int* in_sizes, int n_in,
                              void* d_out, int out_size, void* d_ws, size_t ws_size,
                              hipStream_t stream) {
    const float* x = (const float*)d_in[0];   // [2,3,512,512]
    const float* w = (const float*)d_in[1];   // [180,1,17,17]
    (void)in_sizes; (void)n_in; (void)ws_size; (void)out_size;

    const int npix = NBATCH * IMG_H * IMG_W;  // 524288

    char* ws = (char*)d_ws;
    _Float16* wA = (_Float16*)ws;                        // 208,896 B
    float* G = (float*)(ws + (1 << 18));                 // gray / filtered
    float* T = G + npix;                                 // temp
    float* S = T + npix;                                 // small-sigma blur

    float* outOri  = (float*)d_out;                      // [2,1,512,512]
    float* outConf = outOri + npix;                      // [2,512,512]

    prep_weights_kernel<<<(MT * KH * 32 + 255) / 256, 256, 0, stream>>>(w, wA);
    gray_kernel<<<npix / 256, 256, 0, stream>>>(x, G);
    blurV_kernel<<<npix / 256, 256, 0, stream>>>(G, T, 2, 0.4f);
    blurH_kernel<<<npix / 256, 256, 0, stream>>>(T, S, 2, 0.4f, 0, nullptr);
    blurV_kernel<<<npix / 256, 256, 0, stream>>>(G, T, 40, 10.0f);
    blurH_kernel<<<npix / 256, 256, 0, stream>>>(T, G, 40, 10.0f, 1, S); // G = DoG

    dim3 grid(IMG_W / TILE_X, IMG_H, NBATCH);
    gabor_wmma_kernel<<<grid, 256, 0, stream>>>(G, wA, outOri, outConf);
}
